// TreeLSTM_86380382257425
// MI455X (gfx1250) — compile-verified, matrix operand order checked by reading, not run on hardware
//
#include <hip/hip_runtime.h>

// ---------------------------------------------------------------------------
// TreeLSTM on MI455X (gfx1250, wave32, WMMA).
//
// Roofline: ~56 GFLOP total vs ~0.7 GB of mandatory HBM traffic after fusing
// the embedding GEMMs into the per-level kernels -> memory bound (~30us at
// 23.3 TB/s). So: bf16 WMMA (v_wmma_f32_16x16x32_bf16, f32 accumulate) for
// all 128-deep reductions, weights converted once to column-major bf16
// (~256KB, lives in WGP$/L2), each h/c row touched once per level.
//
// Workspace layout:
//   [0      .. 128KB) : Wcat_T bf16  (512 cols x 128 k)  = [W_iou | W_f]^T
//   [128KB  .. 256KB) : Ucat_T bf16  (512 cols x 128 k)  = [U_iou | U_f]^T
//   [256KB  .. +N*128*4) : c buffer (f32)
// ---------------------------------------------------------------------------

typedef __attribute__((ext_vector_type(16))) __bf16 v16bf;
typedef __attribute__((ext_vector_type(8)))  __bf16 v8bf;
typedef __attribute__((ext_vector_type(8)))  float  v8f;

#define TL_DEPTH 18
#define TL_HS    128   // hidden == embed size

__device__ __forceinline__ float tl_sigmoid(float x) {
    return 1.0f / (1.0f + expf(-x));
}

// Load one 16x32 bf16 WMMA operand fragment (A or B).
//   A: base = 16 rows x 128 k (row-major, stride 128), rc = row (lane&15)
//   B: base = columns stored contiguously over k (stride 128), rc = col
// ISA layout: lanes 0-15 hold K {0..7, 16..23}; lanes 16-31 hold K {8..15, 24..31}.
__device__ __forceinline__ v16bf tl_load_frag(const __bf16* base, int rc, int k0, int lane) {
    const int koff = (lane & 16) ? 8 : 0;
    const __bf16* p = base + rc * TL_HS + k0 + koff;
    v8bf lo = *reinterpret_cast<const v8bf*>(p);        // 16B load
    v8bf hi = *reinterpret_cast<const v8bf*>(p + 16);   // 16B load
    v16bf r;
#pragma unroll
    for (int i = 0; i < 8; ++i) { r[i] = lo[i]; r[i + 8] = hi[i]; }
    return r;
}

// ---------------------------------------------------------------------------
// Weight prep: [W_iou | W_f] and [U_iou | U_f] -> column-major bf16.
// Wcat[n*128 + k] = W(k, n), n in [0,512): n<384 -> W_iou, else W_f.
// ---------------------------------------------------------------------------
__global__ void tl_prep_weights(const float* __restrict__ Wiou, const float* __restrict__ Wf,
                                const float* __restrict__ Uiou, const float* __restrict__ Uf,
                                __bf16* __restrict__ Wcat, __bf16* __restrict__ Ucat) {
    int idx = blockIdx.x * blockDim.x + threadIdx.x;
    if (idx >= 512 * TL_HS) return;
    int n = idx >> 7;
    int k = idx & (TL_HS - 1);
    float w = (n < 384) ? Wiou[k * 384 + n] : Wf[k * TL_HS + (n - 384)];
    float u = (n < 384) ? Uiou[k * 384 + n] : Uf[k * TL_HS + (n - 384)];
    Wcat[n * TL_HS + k] = (__bf16)w;
    Ucat[n * TL_HS + k] = (__bf16)u;
}

// ---------------------------------------------------------------------------
// One tree level. Block = 256 threads (8 wave32) = one 16-node M-tile.
// Per node (640 pre-activation columns in LDS acc):
//   cols [  0,384): iou  = (emb*m)@W_iou + (hl+hr)@U_iou
//   cols [384,512): flp  = (emb*m)@W_f   + hl@U_f
//   cols [512,640): frp  = (emb*m)@W_f   + hr@U_f
// 40 jobs of 8 WMMAs each, split over the 8 waves (5 jobs/wave).
// ---------------------------------------------------------------------------
__global__ __launch_bounds__(256) void tl_level_kernel(
    const int* __restrict__ x, const int* __restrict__ mask,
    const float* __restrict__ E,
    const float* __restrict__ b_iou_x, const float* __restrict__ b_f_x,
    const float* __restrict__ b_iou_h, const float* __restrict__ b_f_h,
    const __bf16* __restrict__ Wcat, const __bf16* __restrict__ Ucat,
    float* __restrict__ cbuf, float* __restrict__ hout,
    int s0, int num, int leaf)
{
    __shared__ alignas(16) __bf16 sEmb[16 * TL_HS];  // (emb * m) rows
    __shared__ alignas(16) __bf16 sHl [16 * TL_HS];
    __shared__ alignas(16) __bf16 sHr [16 * TL_HS];
    __shared__ alignas(16) __bf16 sHs [16 * TL_HS];  // hl + hr
    __shared__ float sAcc[16 * 640];
    __shared__ float sMask[16];

    const int tid      = threadIdx.x;
    const int tileBase = blockIdx.x * 16;

    // ---- Stage A-side operands into LDS (bf16), fold mask into emb ----
    for (int idx = tid; idx < 16 * TL_HS; idx += 256) {
        int r = idx >> 7, col = idx & (TL_HS - 1);
        int t = tileBase + r;
        float e = 0.f, hl = 0.f, hr = 0.f, mval = 0.f;
        if (t < num) {
            int node = s0 + t;
            int mk   = mask[node];
            mval     = (float)mk;
            int tok  = x[node] * mk;
            e = E[tok * TL_HS + col] * mval;
            if (!leaf) {
                hl = hout[(2 * node + 1) * TL_HS + col];
                hr = hout[(2 * node + 2) * TL_HS + col];
            }
        }
        sEmb[idx] = (__bf16)e;
        sHl[idx]  = (__bf16)hl;
        sHr[idx]  = (__bf16)hr;
        sHs[idx]  = (__bf16)(hl + hr);
        if (col == 0) sMask[r] = mval;
    }
    __syncthreads();

    // ---- WMMA phase ----
    const int wave = tid >> 5;
    const int lane = tid & 31;
    const int cw   = lane & 15;          // row (A) / column (B) within tile
    const int m0   = (lane & 16) ? 8 : 0;

    for (int jj = 0; jj < 5; ++jj) {
        int job = wave + jj * 8;                 // 0..39, wave-uniform
        if (leaf && job >= 24) break;            // leaves: iou only
        const __bf16* a2;
        int colBase, accBase;
        if (job < 24)      { colBase = job * 16;              accBase = colBase;       a2 = sHs; }
        else if (job < 32) { colBase = 384 + (job - 24) * 16; accBase = colBase;       a2 = sHl; }
        else               { colBase = 384 + (job - 32) * 16; accBase = colBase + 128; a2 = sHr; }
        const __bf16* wB = Wcat + colBase * TL_HS;
        const __bf16* uB = Ucat + colBase * TL_HS;

        v8f acc = {0.f, 0.f, 0.f, 0.f, 0.f, 0.f, 0.f, 0.f};
#pragma unroll
        for (int k0 = 0; k0 < TL_HS; k0 += 32) {
            v16bf a = tl_load_frag(sEmb, cw, k0, lane);
            v16bf b = tl_load_frag(wB,   cw, k0, lane);
            acc = __builtin_amdgcn_wmma_f32_16x16x32_bf16(false, a, false, b,
                                                          (short)0, acc, false, false);
        }
        if (!leaf) {
#pragma unroll
            for (int k0 = 0; k0 < TL_HS; k0 += 32) {
                v16bf a = tl_load_frag(a2, cw, k0, lane);
                v16bf b = tl_load_frag(uB, cw, k0, lane);
                acc = __builtin_amdgcn_wmma_f32_16x16x32_bf16(false, a, false, b,
                                                              (short)0, acc, false, false);
            }
        }
        // C/D layout: VGPR j -> row (m0+j), lane -> column (cw)
        int n = accBase + cw;
#pragma unroll
        for (int j = 0; j < 8; ++j) sAcc[(m0 + j) * 640 + n] = acc[j];
    }
    __syncthreads();

    // ---- Elementwise epilogue ----
    for (int idx = tid; idx < 16 * TL_HS; idx += 256) {
        int r = idx >> 7, j = idx & (TL_HS - 1);
        int t = tileBase + r;
        if (t >= num) continue;
        int node = s0 + t;
        float m  = sMask[r];
        const float* arow = sAcc + r * 640;
        float iv = arow[j]         + b_iou_x[j] * m         + b_iou_h[j];
        float ov = arow[128 + j]   + b_iou_x[128 + j] * m   + b_iou_h[128 + j];
        float uv = arow[256 + j]   + b_iou_x[256 + j] * m   + b_iou_h[256 + j];
        float c_agg = 0.f;
        if (!leaf) {
            float flp = arow[384 + j] + b_f_x[j] * m + b_f_h[j];
            float frp = arow[512 + j] + b_f_x[j] * m + b_f_h[j];
            float cl  = cbuf[(2 * node + 1) * TL_HS + j];
            float cr  = cbuf[(2 * node + 2) * TL_HS + j];
            c_agg = tl_sigmoid(flp) * cl + tl_sigmoid(frp) * cr;
        }
        float cn = tl_sigmoid(iv) * tanhf(uv) + c_agg;
        float hn = tl_sigmoid(ov) * tanhf(cn);
        cbuf[node * TL_HS + j] = cn;
        hout[node * TL_HS + j] = hn;
    }
}

// ---------------------------------------------------------------------------
extern "C" void kernel_launch(void* const* d_in, const int* in_sizes, int n_in,
                              void* d_out, int out_size, void* d_ws, size_t ws_size,
                              hipStream_t stream) {
    const int*   x       = (const int*)  d_in[0];
    const int*   mask    = (const int*)  d_in[1];
    const float* E       = (const float*)d_in[2];
    const float* W_iou   = (const float*)d_in[3];
    const float* b_iou_x = (const float*)d_in[4];
    const float* W_f     = (const float*)d_in[5];
    const float* b_f_x   = (const float*)d_in[6];
    const float* U_iou   = (const float*)d_in[7];
    const float* b_iou_h = (const float*)d_in[8];
    const float* U_f     = (const float*)d_in[9];
    const float* b_f_h   = (const float*)d_in[10];

    __bf16* Wcat = (__bf16*)d_ws;                       // 512*128*2 = 128KB
    __bf16* Ucat = Wcat + 512 * TL_HS;                  // 128KB
    float*  cbuf = (float*)((char*)d_ws + 2 * 512 * TL_HS * sizeof(__bf16));
    float*  hout = (float*)d_out;

    tl_prep_weights<<<(512 * TL_HS + 255) / 256, 256, 0, stream>>>(
        W_iou, W_f, U_iou, U_f, Wcat, Ucat);

    for (int lvl = TL_DEPTH - 1; lvl >= 0; --lvl) {
        int s0   = (1 << lvl) - 1;
        int num  = 1 << lvl;
        int nblk = (num + 15) / 16;
        int leaf = (lvl == TL_DEPTH - 1) ? 1 : 0;
        tl_level_kernel<<<nblk, 256, 0, stream>>>(
            x, mask, E, b_iou_x, b_f_x, b_iou_h, b_f_h,
            Wcat, Ucat, cbuf, hout, s0, num, leaf);
    }
}
